// TwoStream_GCN_27101243638195
// MI455X (gfx1250) — compile-verified, matrix operand order checked by reading, not run on hardware
//
#include <hip/hip_runtime.h>
#include <hip/hip_bf16.h>

typedef __attribute__((ext_vector_type(16))) _Float16 v16h;
typedef __attribute__((ext_vector_type(8)))  float    v8f;

// ---------------------------------------------------------------------------
// SpMM: out[src[e], :] += val[e] * x[dst[e], :]   (one wave32 per edge)
// Feature matrices (<=102MB) live in the 192MB L2, so gathers/atomics mostly
// hit L2; hardware global_atomic_add_f32 via unsafeAtomicAdd.
// ---------------------------------------------------------------------------
__global__ void spmm_atomic_kernel(const int* __restrict__ src, const int* __restrict__ dst,
                                   const float* __restrict__ val, const float* __restrict__ x,
                                   float* __restrict__ out, int E, int F) {
  int gid  = blockIdx.x * blockDim.x + threadIdx.x;
  int lane = gid & 31;
  int wv   = gid >> 5;
  int nwv  = (gridDim.x * blockDim.x) >> 5;
  int nf4  = F >> 2;
  for (int e = wv; e < E; e += nwv) {
    int s = src[e];
    int d = dst[e];
    float v = val[e];
    const float4* xr = (const float4*)(x + (size_t)d * F);
    float* orow = out + (size_t)s * F;
    for (int f4 = lane; f4 < nf4; f4 += 32) {
      float4 xv = xr[f4];
      int fo = f4 << 2;
      unsafeAtomicAdd(orow + fo + 0, v * xv.x);
      unsafeAtomicAdd(orow + fo + 1, v * xv.y);
      unsafeAtomicAdd(orow + fo + 2, v * xv.z);
      unsafeAtomicAdd(orow + fo + 3, v * xv.w);
    }
  }
}

// ---------------------------------------------------------------------------
// WMMA GEMM: C[M,Nc] (leading dim ldc) = act(A[M,K] @ W[K,Nc] + bias)
// f32 -> f16 operands, f32 accumulation, v_wmma_f32_16x16x32_f16.
// Block = 256 threads = 8 wave32. Each wave owns a 16-row x 64-col tile
// (TN=4 WMMA column tiles reusing one A fragment -> 4x less A traffic).
// Weights staged to LDS in fragment order [tile][kblk][lane][16 halves] so
// each lane reads its v16h B fragment as one contiguous 32B chunk.
// K templated (128/256) -> fully unrolled, branch-free inner loop.
// ---------------------------------------------------------------------------
template <int K>
__global__ void gemm_wmma_kernel(const float* __restrict__ A, const float* __restrict__ W,
                                 const float* __restrict__ bias, float* __restrict__ Cout,
                                 int M, int Nc, int ldc, int relu) {
  constexpr int KB = K / 32;
  constexpr int TN = 4;                 // column tiles per wave
  constexpr int TILE_H = K * 16;        // halves per column tile in LDS
  __shared__ __align__(32) _Float16 sW[TN * TILE_H];
  const int tid = threadIdx.x;
  const int colBase = blockIdx.x * (16 * TN);

  // Stage W[:, colBase:colBase+64] into LDS, pre-swizzled into WMMA B-fragment
  // order: element i of lane ln in k-block kb of tile t holds
  // W[k, colBase + t*16 + (ln&15)], k = kb*32 + (ln>=16?8:0) + (i<8 ? i : i+8).
  for (int j = tid; j < TN * TILE_H; j += 256) {
    int t    = j / TILE_H;
    int j2   = j - t * TILE_H;
    int i    = j2 & 15;
    int ln   = (j2 >> 4) & 31;
    int kblk = j2 >> 9;
    int k = kblk * 32 + ((ln >= 16) ? 8 : 0) + ((i < 8) ? i : (i + 8));
    sW[j] = (_Float16)W[(size_t)k * Nc + colBase + t * 16 + (ln & 15)];
  }
  __syncthreads();

  const int wave    = tid >> 5;
  const int lane    = tid & 31;
  const int rowBase = blockIdx.y * 128 + wave * 16;
  const int row     = rowBase + (lane & 15);
  const bool hi     = lane >= 16;

  // Branch-free guard: clamp the row; garbage rows never get stored.
  const int rclamp = (row < M) ? row : (M - 1);
  const float* arow = A + (size_t)rclamp * K;
  const _Float16* bfrag = sW + lane * 16;

  v8f acc[TN] = {};
  #pragma unroll
  for (int kblk = 0; kblk < KB; ++kblk) {
    const int kb0 = kblk * 32 + (hi ? 8 : 0);
    const int kb1 = kb0 + 16;
    float4 a0 = *(const float4*)(arow + kb0);
    float4 a1 = *(const float4*)(arow + kb0 + 4);
    float4 a2 = *(const float4*)(arow + kb1);
    float4 a3 = *(const float4*)(arow + kb1 + 4);
    v16h af;
    af[0]  = (_Float16)a0.x; af[1]  = (_Float16)a0.y; af[2]  = (_Float16)a0.z; af[3]  = (_Float16)a0.w;
    af[4]  = (_Float16)a1.x; af[5]  = (_Float16)a1.y; af[6]  = (_Float16)a1.z; af[7]  = (_Float16)a1.w;
    af[8]  = (_Float16)a2.x; af[9]  = (_Float16)a2.y; af[10] = (_Float16)a2.z; af[11] = (_Float16)a2.w;
    af[12] = (_Float16)a3.x; af[13] = (_Float16)a3.y; af[14] = (_Float16)a3.z; af[15] = (_Float16)a3.w;
    #pragma unroll
    for (int t = 0; t < TN; ++t) {
      v16h bf = *(const v16h*)(bfrag + t * TILE_H + kblk * 32 * 16);
      acc[t] = __builtin_amdgcn_wmma_f32_16x16x32_f16(false, af, false, bf, (short)0, acc[t],
                                                      false, false);
    }
  }

  // C/D layout: lanes 0-15 -> M=rowBase+r, lanes 16-31 -> M=rowBase+8+r; N=lane&15
  const int rb = rowBase + (hi ? 8 : 0);
  #pragma unroll
  for (int t = 0; t < TN; ++t) {
    const int colx = colBase + t * 16 + (lane & 15);
    const float bs = bias ? bias[colx] : 0.0f;
    #pragma unroll
    for (int r = 0; r < 8; ++r) {
      int ro = rb + r;
      if (ro < M) {
        float v = acc[t][r] + bs;
        if (relu) v = fmaxf(v, 0.0f);
        Cout[(size_t)ro * ldc + colx] = v;
      }
    }
  }
}

// ---------------------------------------------------------------------------
// BatchNorm (train mode) helpers
// ---------------------------------------------------------------------------
__global__ void colstats_kernel(const float* __restrict__ Y, int M, int C,
                                float* __restrict__ sums, float* __restrict__ sumsq) {
  int c = threadIdx.x;
  if (c >= C) return;
  float s = 0.0f, q = 0.0f;
  for (int r = blockIdx.x; r < M; r += gridDim.x) {
    float v = Y[(size_t)r * C + c];
    s += v;
    q += v * v;
  }
  unsafeAtomicAdd(&sums[c], s);
  unsafeAtomicAdd(&sumsq[c], q);
}

__global__ void bnfinalize_kernel(const float* __restrict__ sums, const float* __restrict__ sumsq,
                                  const float* __restrict__ g, const float* __restrict__ b,
                                  float* __restrict__ scale, float* __restrict__ shift,
                                  int C, float invM) {
  int c = threadIdx.x;
  if (c >= C) return;
  float mean = sums[c] * invM;
  float var  = sumsq[c] * invM - mean * mean;
  float sc   = g[c] * rsqrtf(var + 1e-3f);
  scale[c] = sc;
  shift[c] = b[c] - mean * sc;
}

__global__ void bn_skip_relu_kernel(const float* __restrict__ Y, const float* __restrict__ scale,
                                    const float* __restrict__ shift, const float* __restrict__ skip,
                                    float* __restrict__ out, size_t total, int cmask) {
  size_t stride = (size_t)gridDim.x * blockDim.x;
  for (size_t i = (size_t)blockIdx.x * blockDim.x + threadIdx.x; i < total; i += stride) {
    int c = (int)i & cmask;
    float v = fmaf(Y[i], scale[c], shift[c]) + skip[i];
    out[i] = fmaxf(v, 0.0f);
  }
}

// ---------------------------------------------------------------------------
// Pair gather: out[p, 0:128]=feat[idx[2p]], out[p,128:256]=feat[idx[2p+1]]
// ---------------------------------------------------------------------------
__global__ void gather_pairs_kernel(const int* __restrict__ idx, const float* __restrict__ feat,
                                    float* __restrict__ out, size_t total) {
  size_t stride = (size_t)gridDim.x * blockDim.x;
  for (size_t i = (size_t)blockIdx.x * blockDim.x + threadIdx.x; i < total; i += stride) {
    size_t p  = i >> 8;          // / 256
    int w     = (int)(i & 255);
    int which = w >> 7;
    int f     = w & 127;
    int node  = idx[2 * p + which];
    out[i] = feat[(size_t)node * 128 + f];
  }
}

// ---------------------------------------------------------------------------
// Final FC: out[P,2] = H[P,128] @ W[128,2] + b
// ---------------------------------------------------------------------------
__global__ void final_fc2_kernel(const float* __restrict__ H, const float* __restrict__ W,
                                 const float* __restrict__ b, float* __restrict__ out, int P) {
  int p = blockIdx.x * blockDim.x + threadIdx.x;
  if (p >= P) return;
  const float* h = H + (size_t)p * 128;
  float a0 = b[0], a1 = b[1];
  #pragma unroll 4
  for (int k = 0; k < 128; ++k) {
    float v = h[k];
    a0 = fmaf(v, W[2 * k + 0], a0);
    a1 = fmaf(v, W[2 * k + 1], a1);
  }
  out[2 * p + 0] = a0;
  out[2 * p + 1] = a1;
}

// ---------------------------------------------------------------------------
// Host orchestration
// ---------------------------------------------------------------------------
static inline void launch_gemm(const float* A, const float* W, const float* bias, float* C,
                               int M, int K, int Nc, int ldc, int relu, hipStream_t s) {
  dim3 grid(Nc / 64, (M + 127) / 128);
  if (K == 128)
    gemm_wmma_kernel<128><<<grid, 256, 0, s>>>(A, W, bias, C, M, Nc, ldc, relu);
  else
    gemm_wmma_kernel<256><<<grid, 256, 0, s>>>(A, W, bias, C, M, Nc, ldc, relu);
}

extern "C" void kernel_launch(void* const* d_in, const int* in_sizes, int n_in,
                              void* d_out, int out_size, void* d_ws, size_t ws_size,
                              hipStream_t stream) {
  const int*   sp_src    = (const int*)d_in[0];
  const int*   sp_dst    = (const int*)d_in[1];
  const float* sp_val    = (const float*)d_in[2];
  const float* sp_feat   = (const float*)d_in[3];
  const int*   t_src     = (const int*)d_in[4];
  const int*   t_dst     = (const int*)d_in[5];
  const float* t_val     = (const float*)d_in[6];
  const float* t_feat    = (const float*)d_in[7];
  const int*   label_idx = (const int*)d_in[8];
  const float* sp1_W = (const float*)d_in[9],  *sp1_g = (const float*)d_in[10], *sp1_b = (const float*)d_in[11];
  const float* sp2_W = (const float*)d_in[12], *sp2_skip = (const float*)d_in[13],
             * sp2_g = (const float*)d_in[14], *sp2_b = (const float*)d_in[15];
  const float* t1_W  = (const float*)d_in[16], *t1_skip = (const float*)d_in[17],
             * t1_g  = (const float*)d_in[18], *t1_b = (const float*)d_in[19];
  const float* t2_W  = (const float*)d_in[20], *t2_skip = (const float*)d_in[21],
             * t2_g  = (const float*)d_in[22], *t2_b = (const float*)d_in[23];
  const float* scls_W1 = (const float*)d_in[24], *scls_b1 = (const float*)d_in[25];
  const float* scls_W2 = (const float*)d_in[26], *scls_b2 = (const float*)d_in[27];
  const float* tcls_W1 = (const float*)d_in[28], *tcls_b1 = (const float*)d_in[29];
  const float* tcls_W2 = (const float*)d_in[30], *tcls_b2 = (const float*)d_in[31];
  const float* cls_W1 = (const float*)d_in[32], *cls_b1 = (const float*)d_in[33];
  const float* cls_W2 = (const float*)d_in[34], *cls_b2 = (const float*)d_in[35];
  const float* cls_W3 = (const float*)d_in[36], *cls_b3 = (const float*)d_in[37];

  const int E = in_sizes[0];
  const int N = in_sizes[3] / 256;
  const int P = in_sizes[8] / 2;

  // Workspace layout (floats):
  // [stats 1024][A: N*256 persistent node features][R: 3*P*256]
  float* ws       = (float*)d_ws;
  float* st_sum   = ws;          // 256
  float* st_sq    = ws + 256;    // 256
  float* bn_scale = ws + 512;    // 256
  float* bn_shift = ws + 768;    // 256
  float* A = ws + 1024;
  float* R = A + (size_t)N * 256;
  float* B = R;                        // node-phase scratch (aliases E1)
  float* C = R + (size_t)N * 256;
  float* D = R + 2 * (size_t)N * 256;
  float* E1 = R;                       // pair-phase scratch
  float* E2 = R + (size_t)P * 256;
  float* E3 = R + 2 * (size_t)P * 256;

  auto run_bn = [&](const float* Y, int Cc, const float* g, const float* b) {
    hipMemsetAsync(st_sum, 0, 512 * sizeof(float), stream);  // clears sums + sumsq
    colstats_kernel<<<512, Cc, 0, stream>>>(Y, N, Cc, st_sum, st_sq);
    bnfinalize_kernel<<<1, Cc, 0, stream>>>(st_sum, st_sq, g, b, bn_scale, bn_shift, Cc, 1.0f / (float)N);
  };
  const int EW_GRID = 2048, EW_BLK = 256;
  const int SPMM_GRID = 4096;

  // ---- Spatial layer 1: relu(bn((Ahat@x)@W) + x) ----
  hipMemsetAsync(A, 0, (size_t)N * 256 * sizeof(float), stream);
  spmm_atomic_kernel<<<SPMM_GRID, 256, 0, stream>>>(sp_src, sp_dst, sp_val, sp_feat, A, E, 256);
  launch_gemm(A, sp1_W, nullptr, B, N, 256, 256, 256, 0, stream);
  run_bn(B, 256, sp1_g, sp1_b);
  bn_skip_relu_kernel<<<EW_GRID, EW_BLK, 0, stream>>>(B, bn_scale, bn_shift, sp_feat, A,
                                                      (size_t)N * 256, 255);   // s1 -> A

  // ---- Spatial layer 2: relu(bn((Ahat@s1)@W) + s1@Wskip) ----
  hipMemsetAsync(B, 0, (size_t)N * 256 * sizeof(float), stream);
  spmm_atomic_kernel<<<SPMM_GRID, 256, 0, stream>>>(sp_src, sp_dst, sp_val, A, B, E, 256);
  launch_gemm(B, sp2_W, nullptr, C, N, 256, 128, 128, 0, stream);              // Y2
  launch_gemm(A, sp2_skip, nullptr, C + (size_t)N * 128, N, 256, 128, 128, 0, stream);  // skip
  run_bn(C, 128, sp2_g, sp2_b);
  bn_skip_relu_kernel<<<EW_GRID, EW_BLK, 0, stream>>>(C, bn_scale, bn_shift, C + (size_t)N * 128,
                                                      A, (size_t)N * 128, 127);  // s_final -> A[0:N*128]

  // ---- Temporal layer 1 ----
  hipMemsetAsync(B, 0, (size_t)N * 128 * sizeof(float), stream);
  spmm_atomic_kernel<<<SPMM_GRID, 256, 0, stream>>>(t_src, t_dst, t_val, t_feat, B, E, 128);
  launch_gemm(B, t1_W, nullptr, C, N, 128, 256, 256, 0, stream);               // Y
  launch_gemm(t_feat, t1_skip, nullptr, D, N, 128, 256, 256, 0, stream);       // skip
  run_bn(C, 256, t1_g, t1_b);
  bn_skip_relu_kernel<<<EW_GRID, EW_BLK, 0, stream>>>(C, bn_scale, bn_shift, D, B,
                                                      (size_t)N * 256, 255);   // t1_act -> B

  // ---- Temporal layer 2 ----
  hipMemsetAsync(C, 0, (size_t)N * 256 * sizeof(float), stream);
  spmm_atomic_kernel<<<SPMM_GRID, 256, 0, stream>>>(t_src, t_dst, t_val, B, C, E, 256);
  launch_gemm(C, t2_W, nullptr, D, N, 256, 128, 128, 0, stream);               // Y2t
  launch_gemm(B, t2_skip, nullptr, D + (size_t)N * 128, N, 256, 128, 128, 0, stream);
  run_bn(D, 128, t2_g, t2_b);
  bn_skip_relu_kernel<<<EW_GRID, EW_BLK, 0, stream>>>(D, bn_scale, bn_shift, D + (size_t)N * 128,
                                                      A + (size_t)N * 128, (size_t)N * 128, 127);
  // t_final -> A[N*128 : 2*N*128]

  // ---- Spatial classifier MLP over gathered pairs ----
  gather_pairs_kernel<<<EW_GRID, EW_BLK, 0, stream>>>(label_idx, A, E1, (size_t)P * 256);
  launch_gemm(E1, scls_W1, scls_b1, E2, P, 256, 256, 256, 1, stream);
  launch_gemm(E2, scls_W2, scls_b2, E3, P, 256, 128, 256, 0, stream);          // s_pred -> E3[:,0:128]

  // ---- Temporal classifier MLP ----
  gather_pairs_kernel<<<EW_GRID, EW_BLK, 0, stream>>>(label_idx, A + (size_t)N * 128, E1, (size_t)P * 256);
  launch_gemm(E1, tcls_W1, tcls_b1, E2, P, 256, 256, 256, 1, stream);
  launch_gemm(E2, tcls_W2, tcls_b2, E3 + 128, P, 256, 128, 256, 0, stream);    // t_pred -> E3[:,128:256]

  // ---- Final classifier ----
  launch_gemm(E3, cls_W1, cls_b1, E1, P, 256, 256, 256, 1, stream);
  launch_gemm(E1, cls_W2, cls_b2, E2, P, 256, 128, 128, 1, stream);
  final_fc2_kernel<<<(P + 255) / 256, 256, 0, stream>>>(E2, cls_W3, cls_b3, (float*)d_out, P);
}
